// StructuredSelfAttention_39668317945895
// MI455X (gfx1250) — matrix operand order repaired and checked by reading, not compiled, not run on hardware
//
#include <hip/hip_runtime.h>

typedef unsigned short u16;
typedef __attribute__((ext_vector_type(16))) __bf16        v16bf;
typedef __attribute__((ext_vector_type(8)))  float         v8f;
typedef __attribute__((ext_vector_type(4)))  unsigned int  v4u;

union Frag16 {
    v16bf v;
    v4u   q[2];
    u16   u[16];
};

#define BB 64
#define LL 512
#define HH 512
#define DAA 256
#define RR 16
#define DD 64

__device__ __forceinline__ u16 f2bf(float f) {
    unsigned int u = __float_as_uint(f);
    unsigned int r = (u + 0x7fffu + ((u >> 16) & 1u)) >> 16;
    return (u16)r;
}
__device__ __forceinline__ float sigmf(float x) { return 1.0f / (1.0f + __expf(-x)); }

// ---------------------------------------------------------------- grid sync
__device__ __forceinline__ void grid_sync(int* bar, int nblk, int epoch) {
    __syncthreads();
    if (threadIdx.x == 0) {
        __threadfence();
        int ticket = __hip_atomic_fetch_add(&bar[0], 1, __ATOMIC_ACQ_REL, __HIP_MEMORY_SCOPE_AGENT);
        if (ticket == nblk - 1) {
            __hip_atomic_store(&bar[0], 0, __ATOMIC_RELAXED, __HIP_MEMORY_SCOPE_AGENT);
            __hip_atomic_store(&bar[1], epoch, __ATOMIC_RELEASE, __HIP_MEMORY_SCOPE_AGENT);
        } else {
            while (__hip_atomic_load(&bar[1], __ATOMIC_ACQUIRE, __HIP_MEMORY_SCOPE_AGENT) < epoch)
                __builtin_amdgcn_s_sleep(2);
        }
    }
    __syncthreads();
}

// ---------------------------------------------------------------- prep
__global__ void embed_kernel(const int* __restrict__ x, const float* __restrict__ emb,
                             float* __restrict__ ef, u16* __restrict__ eb) {
    int bl = blockIdx.x;           // B*L
    int d  = threadIdx.x;          // 64
    int tok = x[bl];
    float v = emb[(size_t)tok * DD + d];
    ef[(size_t)bl * DD + d] = v;
    eb[(size_t)bl * DD + d] = f2bf(v);
}

__global__ void wprep_kernel(const float* __restrict__ Wih, const float* __restrict__ Whh,
                             const float* __restrict__ bih, const float* __restrict__ bhh,
                             u16* __restrict__ Wc, float* __restrict__ bc, int Kx) {
    int Ktot = Kx + HH;
    size_t idx = (size_t)blockIdx.x * 256 + threadIdx.x;
    size_t total = (size_t)(4 * HH) * Ktot;
    if (idx >= total) return;
    int n = (int)(idx / Ktot), k = (int)(idx % Ktot);
    float v = (k < Kx) ? Wih[(size_t)n * Kx + k] : Whh[(size_t)n * HH + (k - Kx)];
    Wc[idx] = f2bf(v);
    if (k == 0) bc[n] = bih[n] + bhh[n];
}

__global__ void f2bf_kernel(const float* __restrict__ src, u16* __restrict__ dst, int n) {
    int i = blockIdx.x * 256 + threadIdx.x;
    if (i < n) dst[i] = f2bf(src[i]);
}

// ---------------------------------------------------------------- LSTM scan (persistent)
// 32 blocks x 128 threads = 128 waves. wave -> (m_tile 0..3, n_tile 0..31).
// Each wave: all 4 gate tiles for its 16x16 (batch x hidden) patch; c-state in VGPRs.
// K-loop is split into a straight x-part and h-part (no per-chunk branch); all 4
// gate B-fragments are loaded before the WMMA chain so loads clause and overlap.
__global__ __launch_bounds__(128, 1)
void lstm_scan_kernel(const u16* __restrict__ xseq,      // [B, L, Kx] bf16
                      const u16* __restrict__ Wcomb,     // [2048, Kx+512] bf16
                      const float* __restrict__ biascomb,// [2048]
                      u16* __restrict__ hstate,          // [2, 64, 512] bf16 ping-pong
                      u16* __restrict__ outseq_bf,       // [B, L, 512] bf16
                      float* __restrict__ outseq_f,      // [B, L, 512] or null
                      int Kx, int* bar) {
    const int Ktot = Kx + HH;
    const int lane = threadIdx.x & 31;
    const int wid  = blockIdx.x * 4 + (threadIdx.x >> 5);
    const int mt = wid >> 5;          // 0..3
    const int nt = wid & 31;          // 0..31
    const int row0 = mt * 16;
    const int lm = lane & 15;
    const int hi = lane >> 4;
    const int brow = row0 + lm;
    const int nkx = Kx >> 5;

    // loop-invariant lane base pointers
    const u16* xrow = xseq + (size_t)brow * LL * Kx + hi * 8;        // + t*Kx per step
    const u16* hrow0 = hstate + (size_t)brow * HH + hi * 8;          // + parity*BB*HH
    const u16* wb[4];
    #pragma unroll
    for (int g = 0; g < 4; ++g)
        wb[g] = Wcomb + (size_t)(g * HH + nt * 16 + lm) * Ktot + hi * 8;
    const int ncol = nt * 16 + lm;

    v8f cst;
    #pragma unroll
    for (int r = 0; r < 8; ++r) cst[r] = 0.0f;

    float bgate[4];
    #pragma unroll
    for (int g = 0; g < 4; ++g) bgate[g] = biascomb[g * HH + ncol];

    for (int t = 0; t < LL; ++t) {
        const u16* hprev = hrow0 + (size_t)(t & 1) * (BB * HH);
        u16*       hnext = hstate + (size_t)((t + 1) & 1) * (BB * HH);

        v8f acc[4];
        #pragma unroll
        for (int g = 0; g < 4; ++g)
            #pragma unroll
            for (int r = 0; r < 8; ++r) acc[g][r] = 0.0f;

        Frag16 a, bfr[4];

        // ---- x part: K = 0 .. Kx
        const u16* xp = xrow + (size_t)t * Kx;
        for (int kc = 0; kc < nkx; ++kc) {
            a.q[0] = *(const v4u*)(xp + kc * 32);
            a.q[1] = *(const v4u*)(xp + kc * 32 + 16);
            #pragma unroll
            for (int g = 0; g < 4; ++g) {
                const u16* bp = wb[g] + kc * 32;
                bfr[g].q[0] = *(const v4u*)(bp);
                bfr[g].q[1] = *(const v4u*)(bp + 16);
            }
            #pragma unroll
            for (int g = 0; g < 4; ++g)
                acc[g] = __builtin_amdgcn_wmma_f32_16x16x32_bf16(
                    false, a.v, false, bfr[g].v, (short)0, acc[g], false, false);
        }

        // ---- h part: K = Kx .. Kx+512  (16 chunks, unroll 2 for load/WMMA overlap)
        #pragma unroll 2
        for (int kc = 0; kc < (HH >> 5); ++kc) {
            a.q[0] = *(const v4u*)(hprev + kc * 32);
            a.q[1] = *(const v4u*)(hprev + kc * 32 + 16);
            #pragma unroll
            for (int g = 0; g < 4; ++g) {
                const u16* bp = wb[g] + (nkx + kc) * 32;
                bfr[g].q[0] = *(const v4u*)(bp);
                bfr[g].q[1] = *(const v4u*)(bp + 16);
            }
            #pragma unroll
            for (int g = 0; g < 4; ++g)
                acc[g] = __builtin_amdgcn_wmma_f32_16x16x32_bf16(
                    false, a.v, false, bfr[g].v, (short)0, acc[g], false, false);
        }

        #pragma unroll
        for (int r = 0; r < 8; ++r) {
            float iv = sigmf(acc[0][r] + bgate[0]);
            float fv = sigmf(acc[1][r] + bgate[1]);
            float gv = tanhf(acc[2][r] + bgate[2]);
            float ov = sigmf(acc[3][r] + bgate[3]);
            float c  = fv * cst[r] + iv * gv;
            cst[r] = c;
            float h = ov * tanhf(c);
            int m = row0 + r + hi * 8;
            hnext[(size_t)m * HH + ncol] = f2bf(h);
            size_t oidx = ((size_t)m * LL + t) * HH + ncol;
            outseq_bf[oidx] = f2bf(h);
            if (outseq_f) outseq_f[oidx] = h;
        }
        __threadfence();
        grid_sync(bar, gridDim.x, t + 1);
    }
}

// ---------------------------------------------------------------- generic WMMA GEMM
// One wave per 16x16 C tile. A [M,K] bf16 row-major.
// b_kn==0: B = W [N,K] row-major (computes A @ W^T) — contiguous fast path.
// b_kn==1: B = [K,N] row-major — strided loads.
__global__ __launch_bounds__(256, 1)
void gemm_wmma_kernel(const u16* __restrict__ A, const u16* __restrict__ B,
                      const float* __restrict__ bias,
                      float* __restrict__ Cf, u16* __restrict__ Cb,
                      int M, int N, int K, int act, int b_kn,
                      long sA, long sB, long sC) {
    const int batch = blockIdx.y;
    A += (size_t)batch * sA;
    B += (size_t)batch * sB;
    const int lane = threadIdx.x & 31;
    const int wid  = blockIdx.x * 8 + (threadIdx.x >> 5);
    const int ntiles = N >> 4;
    const int tiles  = (M >> 4) * ntiles;
    if (wid >= tiles) return;
    const int tm = wid / ntiles, tn = wid % ntiles;
    const int lm = lane & 15, hi = lane >> 4;

    v8f acc;
    #pragma unroll
    for (int r = 0; r < 8; ++r) acc[r] = 0.0f;

    Frag16 a, bf;
    for (int kc = 0; kc < (K >> 5); ++kc) {
        const u16* ap = A + (size_t)(tm * 16 + lm) * K + kc * 32 + hi * 16;
        a.q[0] = *(const v4u*)(ap);
        a.q[1] = *(const v4u*)(ap + 8);
        if (b_kn) {
            const int k0 = kc * 32 + hi * 16;
            const u16* bp = B + (size_t)k0 * N + tn * 16 + lm;
            #pragma unroll
            for (int e = 0; e < 16; ++e) bf.u[e] = bp[(size_t)e * N];
        } else {
            const u16* bp = B + (size_t)(tn * 16 + lm) * K + kc * 32 + hi * 16;
            bf.q[0] = *(const v4u*)(bp);
            bf.q[1] = *(const v4u*)(bp + 8);
        }
        acc = __builtin_amdgcn_wmma_f32_16x16x32_bf16(
            false, a.v, false, bf.v, (short)0, acc, false, false);
    }

    float bv = bias ? bias[tn * 16 + lm] : 0.0f;
    #pragma unroll
    for (int r = 0; r < 8; ++r) {
        float v = acc[r] + bv;
        if (act) v = tanhf(v);
        int m = tm * 16 + r + hi * 8;
        int n = tn * 16 + lm;
        size_t idx = (size_t)batch * sC + (size_t)m * N + n;
        if (Cf) Cf[idx] = v;
        if (Cb) Cb[idx] = f2bf(v);
    }
}

// ---------------------------------------------------------------- LayerNorm over (L,C) per batch
__global__ __launch_bounds__(256, 1)
void ln2d_kernel(const float* __restrict__ x, const float* __restrict__ g,
                 const float* __restrict__ bta,
                 float* __restrict__ outf, u16* __restrict__ outb, int LC) {
    __shared__ float s1[256], s2[256];
    const int b = blockIdx.x;
    const float* xb = x + (size_t)b * LC;
    float s = 0.0f, ss = 0.0f;
    for (int i = threadIdx.x; i < LC; i += 256) { float v = xb[i]; s += v; ss += v * v; }
    s1[threadIdx.x] = s; s2[threadIdx.x] = ss;
    __syncthreads();
    for (int off = 128; off; off >>= 1) {
        if (threadIdx.x < off) { s1[threadIdx.x] += s1[threadIdx.x + off]; s2[threadIdx.x] += s2[threadIdx.x + off]; }
        __syncthreads();
    }
    float mu = s1[0] / LC;
    float var = s2[0] / LC - mu * mu;
    float rstd = rsqrtf(var + 1e-5f);
    for (int i = threadIdx.x; i < LC; i += 256) {
        float v = (xb[i] - mu) * rstd * g[i] + bta[i];
        if (outf) outf[(size_t)b * LC + i] = v;
        if (outb) outb[(size_t)b * LC + i] = f2bf(v);
    }
}

// ---------------------------------------------------------------- softmax over time, emit attention [B,R,L]
__global__ __launch_bounds__(128, 1)
void softmax_time_kernel(const float* __restrict__ t, float* __restrict__ att,
                         u16* __restrict__ attb) {
    __shared__ float sm[128];
    const int b = blockIdx.x / RR, r = blockIdx.x % RR;
    const float* src = t + (size_t)b * LL * RR + r;  // stride RR over l
    float mx = -1e30f;
    for (int l = threadIdx.x; l < LL; l += 128) mx = fmaxf(mx, src[(size_t)l * RR]);
    sm[threadIdx.x] = mx;
    __syncthreads();
    for (int off = 64; off; off >>= 1) {
        if (threadIdx.x < off) sm[threadIdx.x] = fmaxf(sm[threadIdx.x], sm[threadIdx.x + off]);
        __syncthreads();
    }
    float M = sm[0];
    __syncthreads();
    float s = 0.0f;
    for (int l = threadIdx.x; l < LL; l += 128) s += __expf(src[(size_t)l * RR] - M);
    sm[threadIdx.x] = s;
    __syncthreads();
    for (int off = 64; off; off >>= 1) {
        if (threadIdx.x < off) sm[threadIdx.x] += sm[threadIdx.x + off];
        __syncthreads();
    }
    float inv = 1.0f / sm[0];
    for (int l = threadIdx.x; l < LL; l += 128) {
        float v = __expf(src[(size_t)l * RR] - M) * inv;
        size_t o = ((size_t)b * RR + r) * LL + l;
        att[o] = v;
        attb[o] = f2bf(v);
    }
}

// ---------------------------------------------------------------- conv1(sent) + mean over R, fused
__global__ void convavg_kernel(const float* __restrict__ sent, const float* __restrict__ w,
                               const float* __restrict__ bias, float* __restrict__ avg) {
    int idx = blockIdx.x * 256 + threadIdx.x;   // B*H
    if (idx >= BB * HH) return;
    int b = idx >> 9, h = idx & 511;
    float accum = 0.0f;
    for (int o = 0; o < RR; ++o) accum += bias[o];
    for (int i = 0; i < RR; ++i)
        for (int dx = 0; dx < 3; ++dx) {
            int hh = h + dx - 1;
            if (hh < 0 || hh >= HH) continue;
            float ws = 0.0f;
            for (int o = 0; o < RR; ++o) ws += w[((size_t)o * RR + i) * 3 + dx];
            accum += sent[((size_t)b * RR + i) * HH + hh] * ws;
        }
    avg[idx] = accum / (float)RR;
}

// ---------------------------------------------------------------- convinp over e^T -> ma [B,L]
__global__ void convinp_kernel(const float* __restrict__ e, const float* __restrict__ cw,
                               const float* __restrict__ cb, float* __restrict__ ma) {
    int idx = blockIdx.x * 256 + threadIdx.x;   // B*L
    if (idx >= BB * LL) return;
    int b = idx >> 9, l = idx & 511;
    float accum = cb[0];
    for (int dx = 0; dx < 3; ++dx) {
        int lx = l + dx - 1;
        if (lx < 0 || lx >= LL) continue;
        const float* ep = e + ((size_t)b * LL + lx) * DD;
        for (int d = 0; d < DD; ++d) accum += ep[d] * cw[d * 3 + dx];
    }
    ma[idx] = accum;
}

// ---------------------------------------------------------------- final MLP + sigmoid
__global__ __launch_bounds__(32, 1)
void mlp_kernel(const float* __restrict__ ma, const float* __restrict__ avg,
                const float* __restrict__ w1, const float* __restrict__ b1,
                const float* __restrict__ w2, const float* __restrict__ b2,
                float* __restrict__ out) {
    __shared__ float hm[20];
    int b = blockIdx.x, j = threadIdx.x;
    if (j < 20) {
        const float* wr = w1 + (size_t)j * (LL + HH);
        float s = b1[j];
        for (int k = 0; k < LL; ++k) s += ma[(size_t)b * LL + k] * wr[k];
        for (int k = 0; k < HH; ++k) s += avg[(size_t)b * HH + k] * wr[LL + k];
        hm[j] = tanhf(s);
    }
    __syncthreads();
    if (j == 0) {
        float s = b2[0];
        for (int q = 0; q < 20; ++q) s += hm[q] * w2[q];
        out[b] = 1.0f / (1.0f + __expf(-s));
    }
}

// ================================================================ launch
extern "C" void kernel_launch(void* const* d_in, const int* in_sizes, int n_in,
                              void* d_out, int out_size, void* d_ws, size_t ws_size,
                              hipStream_t stream) {
    (void)in_sizes; (void)n_in; (void)out_size; (void)ws_size;

    const int*   x      = (const int*)d_in[0];
    const float* emb    = (const float*)d_in[1];
    const float* Wih[3] = {(const float*)d_in[2], (const float*)d_in[6], (const float*)d_in[10]};
    const float* Whh[3] = {(const float*)d_in[3], (const float*)d_in[7], (const float*)d_in[11]};
    const float* bih[3] = {(const float*)d_in[4], (const float*)d_in[8], (const float*)d_in[12]};
    const float* bhh[3] = {(const float*)d_in[5], (const float*)d_in[9], (const float*)d_in[13]};
    const float* W1 = (const float*)d_in[14]; const float* b1 = (const float*)d_in[15];
    const float* W2 = (const float*)d_in[16]; const float* b2 = (const float*)d_in[17];
    const float* ln1g = (const float*)d_in[18]; const float* ln1b = (const float*)d_in[19];
    const float* ln2g = (const float*)d_in[20]; const float* ln2b = (const float*)d_in[21];
    const float* ln3g = (const float*)d_in[22]; const float* ln3b = (const float*)d_in[23];
    const float* c1w = (const float*)d_in[24]; const float* c1b = (const float*)d_in[25];
    const float* ciw = (const float*)d_in[26]; const float* cib = (const float*)d_in[27];
    const float* m1w = (const float*)d_in[28]; const float* m1b = (const float*)d_in[29];
    const float* m2w = (const float*)d_in[30]; const float* m2b = (const float*)d_in[31];
    float* out = (float*)d_out;

    char* w = (char*)d_ws;
    size_t off = 0;
    auto alloc = [&](size_t bytes) { size_t o = off; off += (bytes + 255) & ~(size_t)255; return o; };

    const size_t BL = (size_t)BB * LL;
    float* e_f32    = (float*)(w + alloc(BL * DD * 4));
    u16*   e_bf     = (u16*)  (w + alloc(BL * DD * 2));
    u16*   Wc0      = (u16*)  (w + alloc((size_t)4 * HH * (DD + HH) * 2));
    u16*   Wc1      = (u16*)  (w + alloc((size_t)4 * HH * (HH + HH) * 2));
    u16*   Wc2      = (u16*)  (w + alloc((size_t)4 * HH * (HH + HH) * 2));
    float* bcomb    = (float*)(w + alloc((size_t)3 * 4 * HH * 4));
    u16*   seq0     = (u16*)  (w + alloc(BL * HH * 2));
    u16*   seq1     = (u16*)  (w + alloc(BL * HH * 2));
    u16*   seq2     = (u16*)  (w + alloc(BL * HH * 2));
    float* out_f32  = (float*)(w + alloc(BL * HH * 4));
    u16*   hstate   = (u16*)  (w + alloc((size_t)2 * BB * HH * 2));
    int*   bar      = (int*)  (w + alloc(3 * 32));
    u16*   ln1_bf   = (u16*)  (w + alloc(BL * HH * 2));
    float* t1_f32   = (float*)(w + alloc(BL * DAA * 4));
    u16*   ln2_bf   = (u16*)  (w + alloc(BL * DAA * 2));
    float* t2_f32   = (float*)(w + alloc(BL * RR * 4));
    float* t2n_f32  = (float*)(w + alloc(BL * RR * 4));
    u16*   att_bf   = (u16*)  (w + alloc((size_t)BB * RR * LL * 2));
    u16*   W1_bf    = (u16*)  (w + alloc((size_t)DAA * HH * 2));
    u16*   W2_bf    = (u16*)  (w + alloc((size_t)RR * DAA * 2));
    float* sent_f   = (float*)(w + alloc((size_t)BB * RR * HH * 4));
    float* avg      = (float*)(w + alloc((size_t)BB * HH * 4));
    float* ma       = (float*)(w + alloc((size_t)BB * LL * 4));

    hipMemsetAsync(bar, 0, 3 * 32, stream);

    // embedding + bf16 copy
    embed_kernel<<<(int)BL, DD, 0, stream>>>(x, emb, e_f32, e_bf);

    // combined weights + bias (bf16)
    {
        size_t t0 = (size_t)4 * HH * (DD + HH);
        wprep_kernel<<<(int)((t0 + 255) / 256), 256, 0, stream>>>(Wih[0], Whh[0], bih[0], bhh[0], Wc0, bcomb, DD);
        size_t t1 = (size_t)4 * HH * (2 * HH);
        wprep_kernel<<<(int)((t1 + 255) / 256), 256, 0, stream>>>(Wih[1], Whh[1], bih[1], bhh[1], Wc1, bcomb + 4 * HH, HH);
        wprep_kernel<<<(int)((t1 + 255) / 256), 256, 0, stream>>>(Wih[2], Whh[2], bih[2], bhh[2], Wc2, bcomb + 8 * HH, HH);
    }
    f2bf_kernel<<<(DAA * HH + 255) / 256, 256, 0, stream>>>(W1, W1_bf, DAA * HH);
    f2bf_kernel<<<(RR * DAA + 255) / 256, 256, 0, stream>>>(W2, W2_bf, RR * DAA);

    // 3 LSTM layers, persistent scan each
    hipMemsetAsync(hstate, 0, (size_t)2 * BB * HH * 2, stream);
    lstm_scan_kernel<<<32, 128, 0, stream>>>(e_bf, Wc0, bcomb, hstate, seq0, nullptr, DD, bar);
    hipMemsetAsync(hstate, 0, (size_t)2 * BB * HH * 2, stream);
    lstm_scan_kernel<<<32, 128, 0, stream>>>(seq0, Wc1, bcomb + 4 * HH, hstate, seq1, nullptr, HH, bar + 8);
    hipMemsetAsync(hstate, 0, (size_t)2 * BB * HH * 2, stream);
    lstm_scan_kernel<<<32, 128, 0, stream>>>(seq1, Wc2, bcomb + 8 * HH, hstate, seq2, out_f32, HH, bar + 16);

    // attention head
    ln2d_kernel<<<BB, 256, 0, stream>>>(out_f32, ln1g, ln1b, nullptr, ln1_bf, LL * HH);
    gemm_wmma_kernel<<<dim3(4096, 1), 256, 0, stream>>>(ln1_bf, W1_bf, b1, t1_f32, nullptr,
                                                        (int)BL, DAA, HH, 1, 0, 0, 0, 0);
    ln2d_kernel<<<BB, 256, 0, stream>>>(t1_f32, ln2g, ln2b, nullptr, ln2_bf, LL * DAA);
    gemm_wmma_kernel<<<dim3(256, 1), 256, 0, stream>>>(ln2_bf, W2_bf, b2, t2_f32, nullptr,
                                                       (int)BL, RR, DAA, 0, 0, 0, 0, 0);
    ln2d_kernel<<<BB, 256, 0, stream>>>(t2_f32, ln3g, ln3b, t2n_f32, nullptr, LL * RR);
    softmax_time_kernel<<<BB * RR, 128, 0, stream>>>(t2n_f32, out + BB, att_bf);

    // sent = attention @ outputs  (per batch: [16,512] x [512,512], B=[K,N] path)
    gemm_wmma_kernel<<<dim3(4, BB), 256, 0, stream>>>(att_bf, seq2, nullptr, sent_f, nullptr,
                                                      RR, HH, LL, 0, 1,
                                                      (long)RR * LL, (long)LL * HH, (long)RR * HH);

    convavg_kernel<<<(BB * HH + 255) / 256, 256, 0, stream>>>(sent_f, c1w, c1b, avg);
    convinp_kernel<<<(BB * LL + 255) / 256, 256, 0, stream>>>(e_f32, ciw, cib, ma);
    mlp_kernel<<<BB, 32, 0, stream>>>(ma, avg, m1w, m1b, m2w, m2b, out);
}
